// CrossMultiHeadedAttention_23347442221586
// MI455X (gfx1250) — matrix operand order repaired and checked by reading, not compiled, not run on hardware
//
#include <hip/hip_runtime.h>
#include <hip/hip_bf16.h>
#include <stdint.h>

typedef __attribute__((ext_vector_type(16))) __bf16 v16bf;
typedef __attribute__((ext_vector_type(8)))  __bf16 v8bf;
typedef __attribute__((ext_vector_type(4)))  __bf16 v4bf;
typedef __attribute__((ext_vector_type(8)))  float  v8f;

#define B_  4
#define S_  2048
#define H_  512
#define NH_ 8
#define D_  64

// ---------------- CDNA5 primitives ----------------

__device__ __forceinline__ v8f wmma_bf16(v16bf a, v16bf b, v8f c) {
  return __builtin_amdgcn_wmma_f32_16x16x32_bf16(
      false, a, false, b, (short)0, c, false, false);
}

// Async DMA: 16B global -> LDS, tracked by ASYNCcnt (gfx1250).
__device__ __forceinline__ void async_b128(uint32_t lds_off, const void* gaddr) {
  asm volatile("global_load_async_to_lds_b128 %0, %1, off"
               :: "v"(lds_off), "v"(gaddr) : "memory");
}
__device__ __forceinline__ void wait_async() {
  asm volatile("s_wait_asynccnt 0x0" ::: "memory");
}
__device__ __forceinline__ uint32_t lds_off(const void* p) {
  return (uint32_t)(size_t)p;
}

// A fragment from row-major [16 x >=32] bf16 tile (ld multiple of 8):
// two contiguous 8-elem K runs per lane -> 2x ds_load_b128.
__device__ __forceinline__ v16bf load_A(const __bf16* p, int ld) {
  const int lane  = threadIdx.x & 31;
  const int row   = lane & 15;
  const int khalf = lane >> 4;
  const __bf16* base = p + row * ld + khalf * 8;
  v8bf lo = *(const v8bf*)(base);
  v8bf hi = *(const v8bf*)(base + 16);
  return __builtin_shufflevector(lo, hi, 0, 1, 2, 3, 4, 5, 6, 7,
                                 8, 9, 10, 11, 12, 13, 14, 15);
}

// B fragment from row-major [N][K] tile holding B^T (ld multiple of 8):
// two contiguous 8-elem K runs per lane -> 2x ds_load_b128.
__device__ __forceinline__ v16bf load_Bt(const __bf16* p, int ld) {
  const int lane  = threadIdx.x & 31;
  const int col   = lane & 15;
  const int khalf = lane >> 4;
  const __bf16* base = p + col * ld + khalf * 8;
  v8bf lo = *(const v8bf*)(base);
  v8bf hi = *(const v8bf*)(base + 16);
  return __builtin_shufflevector(lo, hi, 0, 1, 2, 3, 4, 5, 6, 7,
                                 8, 9, 10, 11, 12, 13, 14, 15);
}

__device__ __forceinline__ void store_bf4(__bf16* dst, float4 x) {
  v4bf t;
  t[0] = (__bf16)x.x; t[1] = (__bf16)x.y; t[2] = (__bf16)x.z; t[3] = (__bf16)x.w;
  *(v4bf*)dst = t;
}

// X-tile fill: fp32 input (convert) or bf16 input (straight 16B copies).
__device__ __forceinline__ void tileA_fill(__bf16 (*sA)[40], const float* X,
                                           int K, int m0, int k0, int tid) {
  for (int i = tid * 4; i < 64 * 32; i += 128 * 4) {
    int r = i >> 5, c = i & 31;
    float4 x = *(const float4*)&X[(size_t)(m0 + r) * K + k0 + c];
    store_bf4(&sA[r][c], x);
  }
}
__device__ __forceinline__ void tileA_fill(__bf16 (*sA)[40], const __bf16* X,
                                           int K, int m0, int k0, int tid) {
  for (int i = tid * 8; i < 64 * 32; i += 128 * 8) {
    int r = i >> 5, c = i & 31;
    *(v8bf*)&sA[r][c] = *(const v8bf*)&X[(size_t)(m0 + r) * K + k0 + c];
  }
}

// ---------------- Projection GEMM ----------------
// acc = X1@W1 (+ X2@W2) + bias1 (+ bias2)
// MODE 0: fp32 row-major [M][N]
// MODE 1: bf16 head-major [B,NH,S,D]
// MODE 2: bf16 transposed head-major [B,NH,D,S] (packed 16B stores)
template <typename Tin, int MODE>
__global__ __launch_bounds__(128)
void proj_gemm(const Tin* __restrict__ X1, const float* __restrict__ W1,
               const Tin* __restrict__ X2, const float* __restrict__ W2,
               const float* __restrict__ bias1, const float* __restrict__ bias2,
               void* __restrict__ OutV, int M, int N, int K)
{
  __shared__ __align__(16) __bf16 sA[64][40];   // X tile, row-major [M][K]
  __shared__ __align__(16) __bf16 sBt[64][40];  // W tile transposed: [N][K]
  const int tid  = threadIdx.x;
  const int lane = tid & 31;
  const int wave = tid >> 5;
  const int m0 = blockIdx.x * 64;
  const int n0 = blockIdx.y * 64;

  v8f acc[4] = {};
  const int npass = X2 ? 2 : 1;
  for (int pass = 0; pass < npass; ++pass) {
    const Tin*  Xp = pass ? X2 : X1;
    const float* Wp = pass ? W2 : W1;
    for (int k0 = 0; k0 < K; k0 += 32) {
      __syncthreads();
      tileA_fill(sA, Xp, K, m0, k0, tid);
      for (int i = tid * 4; i < 32 * 64; i += 128 * 4) {
        int r = i >> 6, c = i & 63;
        float4 w = *(const float4*)&Wp[(size_t)(k0 + r) * N + n0 + c];
        sBt[c + 0][r] = (__bf16)w.x;
        sBt[c + 1][r] = (__bf16)w.y;
        sBt[c + 2][r] = (__bf16)w.z;
        sBt[c + 3][r] = (__bf16)w.w;
      }
      __syncthreads();
      v16bf a  = load_A(&sA[wave * 16][0], 40);
      v16bf b0 = load_Bt(&sBt[0][0], 40);
      v16bf b1 = load_Bt(&sBt[16][0], 40);
      v16bf b2 = load_Bt(&sBt[32][0], 40);
      v16bf b3 = load_Bt(&sBt[48][0], 40);
      acc[0] = wmma_bf16(a, b0, acc[0]);
      acc[1] = wmma_bf16(a, b1, acc[1]);
      acc[2] = wmma_bf16(a, b2, acc[2]);
      acc[3] = wmma_bf16(a, b3, acc[3]);
    }
  }

  const int colh = lane & 15;
  const int mb   = (lane >> 4) << 3;
  const int mm0  = m0 + wave * 16 + mb;   // 8 consecutive output rows
#pragma unroll
  for (int n = 0; n < 4; ++n) {
    int nn = n0 + n * 16 + colh;
    float bval = bias1[nn] + (bias2 ? bias2[nn] : 0.f);
    if (MODE == 0) {
      float* O = (float*)OutV;
#pragma unroll
      for (int v = 0; v < 8; ++v)
        O[(size_t)(mm0 + v) * N + nn] = acc[n][v] + bval;
    } else if (MODE == 1) {
      __bf16* O = (__bf16*)OutV;
      int hh = nn / D_, dd = nn % D_;
#pragma unroll
      for (int v = 0; v < 8; ++v) {
        int mm = mm0 + v, bb = mm / S_, ss = mm % S_;
        O[((((size_t)bb * NH_) + hh) * S_ + ss) * D_ + dd] = (__bf16)(acc[n][v] + bval);
      }
    } else {  // MODE 2: [B,NH,D,S], 8 consecutive s -> one 16B store
      __bf16* O = (__bf16*)OutV;
      int hh = nn / D_, dd = nn % D_;
      int bb = mm0 / S_, ss0 = mm0 % S_;
      v8bf pk;
#pragma unroll
      for (int v = 0; v < 8; ++v) pk[v] = (__bf16)(acc[n][v] + bval);
      *(v8bf*)&O[((((size_t)bb * NH_) + hh) * D_ + dd) * S_ + ss0] = pk;
    }
  }
}

// ---------------- Flash attention (bf16 in, async double-buffered) --------

__device__ __forceinline__ void issue_kv_tile(const __bf16* Kt, const __bf16* Vtp,
                                              int kbase, __bf16 (*bK)[72],
                                              __bf16 (*bV)[72], int tid) {
  // K tile: 64x64 contiguous in global.  V^T tile: rows d, stride S.
  for (int i = tid; i < 512; i += 128) {
    int r = i >> 3, c8 = (i & 7) << 3;
    async_b128(lds_off(&bK[r][c8]), Kt + (size_t)r * D_ + c8);
    async_b128(lds_off(&bV[r][c8]), Vtp + (size_t)r * S_ + kbase + c8);
  }
}

__global__ __launch_bounds__(128)
void flash_attn(const __bf16* __restrict__ Q, const __bf16* __restrict__ Keff,
                const __bf16* __restrict__ Vt, const int* __restrict__ mask,
                __bf16* __restrict__ hid)
{
  __shared__ __align__(16) __bf16 sQ[64][72];       // [q][d]
  __shared__ __align__(16) __bf16 sK[2][64][72];    // [key][d]
  __shared__ __align__(16) __bf16 sVt[2][64][72];   // [d][key]
  __shared__ __align__(16) __bf16 sP[4][16][72];    // per-wave P, [q][key]

  const int tid  = threadIdx.x;
  const int lane = tid & 31;
  const int wave = tid >> 5;
  const int qbase = blockIdx.x * 64;
  const int head  = blockIdx.y;
  const int bb    = blockIdx.z;

  const size_t hoff = (((size_t)bb * NH_) + head) * (size_t)S_ * D_;
  const __bf16* Qp   = Q + hoff + (size_t)qbase * D_;
  const __bf16* Kbase = Keff + hoff;
  const __bf16* Vtp  = Vt + (((size_t)bb * NH_) + head) * (size_t)D_ * S_;

  // Q tile: synchronous 16B copies (once)
  for (int i = tid * 8; i < 64 * 64; i += 128 * 8) {
    int r = i >> 6, c = i & 63;
    *(v8bf*)&sQ[r][c] = *(const v8bf*)&Qp[(size_t)r * D_ + c];
  }
  // prefetch tile 0 into buffer 0 (async DMA)
  issue_kv_tile(Kbase, Vtp, 0, sK[0], sVt[0], tid);
  __syncthreads();
  v16bf aQ0 = load_A(&sQ[wave * 16][0], 72);
  v16bf aQ1 = load_A(&sQ[wave * 16][32], 72);

  float mrow[8], lrow[8];
  v8f o[4] = {};
#pragma unroll
  for (int v = 0; v < 8; ++v) { mrow[v] = -1e30f; lrow[v] = 0.f; }

  const int colh = lane & 15;
  const int mb   = (lane >> 4) << 3;

  for (int kt = 0; kt < S_ / 64; ++kt) {
    const int cur = kt & 1;
    const int kbase = kt * 64;
    wait_async();       // our DMA for buf[cur] complete
    __syncthreads();    // all waves' DMA visible; prev compute done reading other buf
    if (kt + 1 < S_ / 64)
      issue_kv_tile(Kbase + (size_t)(kbase + 64) * D_, Vtp, kbase + 64,
                    sK[1 - cur], sVt[1 - cur], tid);

    // key mask straight from L2
    int mv[4];
#pragma unroll
    for (int n = 0; n < 4; ++n)
      mv[n] = mask[(size_t)bb * S_ + kbase + n * 16 + colh];

    // scores = Q @ Keff^T (K-dim = D = 64)
    v8f sc[4] = {};
#pragma unroll
    for (int n = 0; n < 4; ++n) {
      v16bf b0 = load_Bt(&sK[cur][n * 16][0], 72);
      sc[n] = wmma_bf16(aQ0, b0, sc[n]);
      v16bf b1 = load_Bt(&sK[cur][n * 16][32], 72);
      sc[n] = wmma_bf16(aQ1, b1, sc[n]);
    }
#pragma unroll
    for (int n = 0; n < 4; ++n)
#pragma unroll
      for (int v = 0; v < 8; ++v) {
        float x = sc[n][v] * 0.125f;
        sc[n][v] = (mv[n] == 0) ? -1e9f : x;
      }
    // online softmax (rows live across 16-lane groups)
#pragma unroll
    for (int v = 0; v < 8; ++v) {
      float t = fmaxf(fmaxf(sc[0][v], sc[1][v]), fmaxf(sc[2][v], sc[3][v]));
      t = fmaxf(t, __shfl_xor(t, 1, 32));
      t = fmaxf(t, __shfl_xor(t, 2, 32));
      t = fmaxf(t, __shfl_xor(t, 4, 32));
      t = fmaxf(t, __shfl_xor(t, 8, 32));
      float mnew = fmaxf(mrow[v], t);
      float corr = __expf(mrow[v] - mnew);
      mrow[v] = mnew;
      float rs = 0.f;
#pragma unroll
      for (int n = 0; n < 4; ++n) {
        float p = __expf(sc[n][v] - mnew);
        sc[n][v] = p;
        rs += p;
      }
      rs += __shfl_xor(rs, 1, 32);
      rs += __shfl_xor(rs, 2, 32);
      rs += __shfl_xor(rs, 4, 32);
      rs += __shfl_xor(rs, 8, 32);
      lrow[v] = lrow[v] * corr + rs;
#pragma unroll
      for (int n = 0; n < 4; ++n) o[n][v] *= corr;
    }
    // restripe P (C layout -> row-major [q][key])
#pragma unroll
    for (int n = 0; n < 4; ++n)
#pragma unroll
      for (int v = 0; v < 8; ++v)
        sP[wave][mb + v][n * 16 + colh] = (__bf16)sc[n][v];
    __syncthreads();
    // O += P @ V
    v16bf aP0 = load_A(&sP[wave][0][0], 72);
    v16bf aP1 = load_A(&sP[wave][0][32], 72);
#pragma unroll
    for (int n = 0; n < 4; ++n) {
      v16bf bv0 = load_Bt(&sVt[cur][n * 16][0], 72);
      o[n] = wmma_bf16(aP0, bv0, o[n]);
      v16bf bv1 = load_Bt(&sVt[cur][n * 16][32], 72);
      o[n] = wmma_bf16(aP1, bv1, o[n]);
    }
  }

  // normalize, store hid bf16 [B,S,H]
#pragma unroll
  for (int v = 0; v < 8; ++v) {
    float inv = 1.f / lrow[v];
    int mm = qbase + wave * 16 + mb + v;
#pragma unroll
    for (int n = 0; n < 4; ++n) {
      int hh = head * D_ + n * 16 + colh;
      hid[(((size_t)bb * S_) + mm) * H_ + hh] = (__bf16)(o[n][v] * inv);
    }
  }
}

// ---------------- Host entry ----------------

extern "C" void kernel_launch(void* const* d_in, const int* in_sizes, int n_in,
                              void* d_out, int out_size, void* d_ws, size_t ws_size,
                              hipStream_t stream) {
  (void)in_sizes; (void)n_in; (void)out_size; (void)ws_size;
  const float* q    = (const float*)d_in[0];
  const float* k    = (const float*)d_in[1];
  const float* v    = (const float*)d_in[2];
  const float* kb   = (const float*)d_in[3];
  const int*   mask = (const int*)d_in[4];
  const float* Wq   = (const float*)d_in[5];
  const float* bq   = (const float*)d_in[6];
  const float* Wk   = (const float*)d_in[7];
  const float* bk   = (const float*)d_in[8];
  const float* Wv   = (const float*)d_in[9];
  const float* bv   = (const float*)d_in[10];
  const float* Wkb  = (const float*)d_in[11];
  const float* bkb  = (const float*)d_in[12];
  const float* Wo   = (const float*)d_in[13];
  const float* bo   = (const float*)d_in[14];
  float* out = (float*)d_out;

  const size_t P = (size_t)B_ * NH_ * S_ * D_;  // elements per projection
  __bf16* qh  = (__bf16*)d_ws;      // [B,NH,S,D]
  __bf16* kh  = qh + P;             // [B,NH,S,D]  (keff, fused k + k_b)
  __bf16* vT  = kh + P;             // [B,NH,D,S]
  __bf16* hid = vT + P;             // [B,S,H]

  const int M = B_ * S_;
  dim3 blk(128);
  dim3 ggrid(M / 64, H_ / 64);
  proj_gemm<float, 1><<<ggrid, blk, 0, stream>>>(
      q, Wq, (const float*)nullptr, (const float*)nullptr, bq, nullptr,
      qh, M, H_, H_);
  proj_gemm<float, 1><<<ggrid, blk, 0, stream>>>(
      k, Wk, kb, Wkb, bk, bkb, kh, M, H_, H_);               // fused keff
  proj_gemm<float, 2><<<ggrid, blk, 0, stream>>>(
      v, Wv, (const float*)nullptr, (const float*)nullptr, bv, nullptr,
      vT, M, H_, H_);

  dim3 agrid(S_ / 64, NH_, B_);
  flash_attn<<<agrid, blk, 0, stream>>>(qh, kh, vT, mask, hid);

  proj_gemm<__bf16, 0><<<ggrid, blk, 0, stream>>>(
      hid, Wo, (const __bf16*)nullptr, (const float*)nullptr, bo, nullptr,
      out, M, H_, H_);
}
